// ModulatedConv3d_33638183862863
// MI455X (gfx1250) — matrix operand order repaired and checked by reading, not compiled, not run on hardware
//
#include <hip/hip_runtime.h>
#include <stdint.h>

typedef __attribute__((ext_vector_type(16))) __bf16 v16bf;
typedef __attribute__((ext_vector_type(8)))  float  v8f;
typedef __attribute__((ext_vector_type(4)))  int    v4i;

#define CIN   128
#define COUT  128
#define BATCH 4
#define KVOL  27
#define HALO_ROWS  18              // 3 z  x  6 x
#define HALO_Y     34              // y in [-1, 32]
#define HALO_ELEMS (HALO_ROWS * HALO_Y * 32)         // 19584 bf16
#define HALO_CHUNKS (HALO_ROWS * HALO_Y * 4)         // 2448 x 16B
#define MAX_CHUNKS_PER_THREAD 10                     // ceil(2448/256)
#define DUMP_OFF   HALO_ELEMS                        // 16B dump slot for dead lanes

struct U2 { uint4 a, b; };

__device__ __forceinline__ unsigned short f32_to_bf16(float f) {
    unsigned int u = __float_as_uint(f);
    u += 0x7FFFu + ((u >> 16) & 1u);          // round-to-nearest-even
    return (unsigned short)(u >> 16);
}

#if __has_builtin(__builtin_amdgcn_global_load_async_to_lds_b128)
#define HAVE_ASYNC_LDS 1
typedef __attribute__((address_space(1))) v4i* gv4p;   // global int4*
typedef __attribute__((address_space(3))) v4i* lv4p;   // LDS int4*
#else
#define HAVE_ASYNC_LDS 0
#endif

// ---------------------------------------------------------------------------
// Kernel 1: per (b, cout): modulate by (S+1), demodulate by rsqrt(sum w^2+eps),
// write bf16 weights in layout Wbf[b][tap][cout][cin].
// ---------------------------------------------------------------------------
__global__ __launch_bounds__(CIN) void modw_kernel(
    const float* __restrict__ S, const float* __restrict__ W,
    unsigned short* __restrict__ Wbf)
{
    const int b    = blockIdx.x >> 7;
    const int cout = blockIdx.x & 127;
    const int cin  = threadIdx.x;

    const float s = S[b * CIN + cin] + 1.0f;
    const float* wp = W + ((size_t)cout * CIN + cin) * KVOL;

    float w[KVOL];
    float ss = 0.f;
#pragma unroll
    for (int k = 0; k < KVOL; ++k) { w[k] = wp[k] * s; ss += w[k] * w[k]; }

    __shared__ float red[CIN];
    red[cin] = ss;
    __syncthreads();
#pragma unroll
    for (int off = CIN / 2; off > 0; off >>= 1) {
        if (cin < off) red[cin] += red[cin + off];
        __syncthreads();
    }
    const float d = rsqrtf(red[0] + 1e-8f);

#pragma unroll
    for (int k = 0; k < KVOL; ++k) {
        Wbf[((size_t)(b * KVOL + k) * COUT + cout) * CIN + cin] = f32_to_bf16(w[k] * d);
    }
}

// ---------------------------------------------------------------------------
// Kernel 2: transpose x to bf16 channel-last: xbf[b][z][x][y][cin].
// ---------------------------------------------------------------------------
__global__ __launch_bounds__(256) void xpose_kernel(
    const float* __restrict__ x, unsigned short* __restrict__ xbf)
{
    __shared__ unsigned short Ts[32 * 128];   // [y][cin]
    const int tid = threadIdx.x;
    const int blk = blockIdx.x;               // 4096 = b*1024 + (z*32 + x)
    const int b   = blk >> 10;
    const int zx  = blk & 1023;

    const size_t inBase = (((size_t)b * CIN) << 15) + ((size_t)zx << 5);
#pragma unroll
    for (int i = 0; i < 16; ++i) {
        const int e    = tid + (i << 8);
        const int cinr = e >> 5;
        const int y    = e & 31;
        Ts[y * 128 + cinr] = f32_to_bf16(x[inBase + ((size_t)cinr << 15) + y]);
    }
    __syncthreads();
    unsigned short* dst = xbf + (((size_t)b << 10) + zx) * (32 * 128);
#pragma unroll
    for (int i = 0; i < 4; ++i) {
        const int e  = tid + (i << 8);        // 0..1023
        const int y  = e >> 5;
        const int cg = (e & 31) << 2;         // cin (4 at a time)
        *(uint2*)(dst + y * 128 + cg) = *(const uint2*)(&Ts[y * 128 + cg]);
    }
}

// ---------------------------------------------------------------------------
// Kernel 3: implicit GEMM with halo-LDS. Block = 256 threads (8 waves), owns
// one batch, a (z, x0..x0+3, all-y) 128-voxel tile, all 128 couts.
// Wave blocking: 32 couts x 64 voxels -> 1 ds_load_b128 per WMMA.
// Staging: branchless async Global->LDS (dead chunks target a dump slot).
// ---------------------------------------------------------------------------
__global__ __launch_bounds__(256) void conv_kernel(
    const unsigned short* __restrict__ xbf, const unsigned short* __restrict__ Wbf,
    float* __restrict__ out)
{
    __shared__ unsigned short Xs[HALO_ELEMS + 8];   // + 16B dump slot

    const int tid  = threadIdx.x;
    const int lane = tid & 31;
    const int wave = tid >> 5;

    const int b   = blockIdx.x >> 8;            // 1024 blocks = b*256 + tile
    const int rem = blockIdx.x & 255;
    const int z   = rem >> 3;                   // 0..31
    const int x0  = (rem & 7) << 2;             // 0,4,..,28
    const int v0  = (z << 10) + (x0 << 5);

    const int col   = lane & 15;                // N column for B/C frags
    const int kb0   = (lane >> 4) << 3;         // B frag: K base 0 or 8
    const int ka0   = (lane >> 4) << 3;         // A frag: K base 0 or 8
    const int wcout = wave & 3;                 // 4 waves span 128 couts
    const int wvox  = wave >> 2;                // 2 wave-groups span 128 voxels
    const int cout0 = wcout << 5;               // 0,32,64,96

    // per-lane B-frag base, including this wave's voxel-half (2 halo x-rows)
    const unsigned short* XsLane = &Xs[col * 32 + kb0 + wvox * (2 * HALO_Y * 32)];
    const unsigned short* xbfB   = xbf + ((size_t)b << 22);  // batch base

    // ---- hoisted halo-chunk descriptors (identical for all cin-blocks) ----
    int goff[MAX_CHUNKS_PER_THREAD];   // global element offset (sans cin0)
    int loff[MAX_CHUNKS_PER_THREAD];   // LDS element offset
#pragma unroll
    for (int i = 0; i < MAX_CHUNKS_PER_THREAD; ++i) {
        const int e = tid + (i << 8);
        goff[i] = 0; loff[i] = DUMP_OFF;       // default: dead chunk -> dump
        if (e < HALO_CHUNKS) {
            const int c   = e & 3;             // 16B chunk within position
            const int pos = e >> 2;            // 0..611
            const int ri  = pos / 34;          // 0..17
            const int yi  = pos - ri * 34;
            const int zi  = ri / 6;
            const int xi  = ri - zi * 6;
            const int zz  = z  + zi - 1;
            const int xg  = x0 + xi - 1;
            const int yy  = yi - 1;
            const bool inb = ((unsigned)zz < 32u) & ((unsigned)xg < 32u) &
                             ((unsigned)yy < 32u);
            const int lo = (ri * HALO_Y + yi) * 32 + (c << 3);
            if (inb) {
                loff[i] = lo;
                goff[i] = (((zz * 32 + xg) * 32 + yy) << 7) + (c << 3);
            } else {
                // zero-fill once: these LDS slots are never overwritten later
                uint4 zv = {0u, 0u, 0u, 0u};
                *(uint4*)(&Xs[lo]) = zv;
            }
        }
    }

    v8f acc[8] = {};   // [mi*4 + t]: mi = cout half (16), t = voxel subtile (16)

#pragma unroll 1
    for (int cb = 0; cb < 4; ++cb) {
        const int cin0 = cb << 5;
        __syncthreads();   // WAR vs previous compute phase (covers zero-fill too)

        // ---- stage halo chunks: branchless 16B Global -> LDS ----
#pragma unroll
        for (int i = 0; i < MAX_CHUNKS_PER_THREAD; ++i) {
#if HAVE_ASYNC_LDS
            __builtin_amdgcn_global_load_async_to_lds_b128(
                (gv4p)(xbfB + goff[i] + cin0), (lv4p)(&Xs[loff[i]]), 0, 0);
#else
            uint4 v = *(const uint4*)(xbfB + goff[i] + cin0);
            *(uint4*)(&Xs[loff[i]]) = v;
#endif
        }
#if HAVE_ASYNC_LDS
        asm volatile("s_wait_asynccnt 0" ::: "memory");
#endif
        __syncthreads();

        // ---- compute: 27 taps; 2 A frags x 4 B frags -> 8 WMMA per tap ----
        const unsigned short* apBase =
            Wbf + ((size_t)(b * KVOL) * COUT + cout0 + col) * CIN + cin0 + ka0;
#pragma unroll
        for (int ko = 0; ko < KVOL; ++ko) {
            const int dz1 = ko / 9;          // dz + 1
            const int dx1 = (ko / 3) % 3;    // dx + 1
            const int dy1 = ko % 3;          // dy + 1

            const unsigned short* ap = apBase + (size_t)ko * (COUT * CIN);
            U2 au0, au1;
            au0.a = *(const uint4*)(ap);                  // rows cout0+col
            au0.b = *(const uint4*)(ap + 16);
            au1.a = *(const uint4*)(ap + 16 * CIN);       // rows cout0+16+col
            au1.b = *(const uint4*)(ap + 16 * CIN + 16);
            const v16bf a0 = __builtin_bit_cast(v16bf, au0);
            const v16bf a1 = __builtin_bit_cast(v16bf, au1);

#pragma unroll
            for (int t = 0; t < 4; ++t) {
                // halo row = dz1*6 + dx1 + (t>>1), y = (t&1)*16 + col + dy1
                const int koff =
                    ((dz1 * 6 + dx1 + (t >> 1)) * HALO_Y + (t & 1) * 16 + dy1) * 32;
                U2 bu;
                bu.a = *(const uint4*)(XsLane + koff);      // K = kb0..kb0+7
                bu.b = *(const uint4*)(XsLane + koff + 16); // K = kb0+16..kb0+23
                const v16bf bfrag = __builtin_bit_cast(v16bf, bu);
                acc[t] = __builtin_amdgcn_wmma_f32_16x16x32_bf16(
                    false, a0, false, bfrag, (short)0, acc[t], false, false);
                acc[4 + t] = __builtin_amdgcn_wmma_f32_16x16x32_bf16(
                    false, a1, false, bfrag, (short)0, acc[4 + t], false, false);
            }
        }
    }

    // ---- epilogue: C layout -> VGPR r holds M = r + 8*(lane>=16), N = col ----
    const int rowoff = (lane >> 4) << 3;
#pragma unroll
    for (int mi = 0; mi < 2; ++mi) {
#pragma unroll
        for (int t = 0; t < 4; ++t) {
#pragma unroll
            for (int r = 0; r < 8; ++r) {
                out[((size_t)(b * COUT + cout0 + (mi << 4) + rowoff + r) << 15) +
                    v0 + (wvox << 6) + (t << 4) + col] = acc[mi * 4 + t][r];
            }
        }
    }
}

extern "C" void kernel_launch(void* const* d_in, const int* in_sizes, int n_in,
                              void* d_out, int out_size, void* d_ws, size_t ws_size,
                              hipStream_t stream) {
    const float* x = (const float*)d_in[0];   // [4,128,32,32,32]
    const float* S = (const float*)d_in[1];   // [4,128]
    const float* W = (const float*)d_in[2];   // [128,128,3,3,3]
    float* out = (float*)d_out;               // [4,128,32,32,32]

    unsigned short* Wbf = (unsigned short*)d_ws;                       // 3.54 MB
    unsigned short* xbf = (unsigned short*)((char*)d_ws + (4u << 20)); // 33.5 MB

    modw_kernel<<<BATCH * COUT, CIN, 0, stream>>>(S, W, Wbf);
    xpose_kernel<<<BATCH * 1024, 256, 0, stream>>>(x, xbf);
    conv_kernel<<<BATCH * 256, 256, 0, stream>>>(xbf, Wbf, out);
}
